// BatchSplitFF_85083302133888
// MI455X (gfx1250) — compile-verified, matrix operand order checked by reading, not run on hardware
//
#include <hip/hip_runtime.h>

// Problem constants (from reference)
#define DM      1024
#define NE      32
#define NSETS   4
#define ESP     128          // NE * NSETS
#define G       32           // tokens per group (= NEXPERTS)
#define BB      2
#define SS      8192
#define NGR     256          // SS / G
#define NGROUPS 512          // BB * NGR

typedef __attribute__((ext_vector_type(16))) __bf16 v16bf;
typedef __attribute__((ext_vector_type(8)))  __bf16 v8bf;
typedef __attribute__((ext_vector_type(8)))  float  v8f;

// Load a 16-element bf16 WMMA operand fragment from an LDS row.
// Layout (ISA 7.12.2, 16-bit A/B): lanes<16 hold K = {off..off+7, off+16..off+23},
// lanes>=16 hold K shifted by +8. Caller passes off already including the +8
// for high lanes, so this is two aligned 16-byte ds_load_b128s.
__device__ __forceinline__ v16bf frag16(const __bf16* row, int off) {
  union { v16bf v; v8bf h[2]; } u;
  u.h[0] = *(const v8bf*)(row + off);
  u.h[1] = *(const v8bf*)(row + off + 16);
  return u.v;
}

__device__ __forceinline__ v8f wmma_bf16(v16bf a, v16bf b, v8f c) {
  return __builtin_amdgcn_wmma_f32_16x16x32_bf16(false, a, false, b, (short)0, c,
                                                 false, false);
}

// Packed f32 -> bf16 LDS stores (one b32 / b64 per 2 / 4 elements).
__device__ __forceinline__ void store_pk2(__bf16* dst, float a, float b) {
  union { __bf16 h[2]; unsigned u; } p;
  p.h[0] = (__bf16)a; p.h[1] = (__bf16)b;
  *(unsigned*)dst = p.u;                       // dst 4B aligned
}
__device__ __forceinline__ void store_pk4(__bf16* dst, float4 v) {
  union { __bf16 h[4]; uint2 u; } p;
  p.h[0] = (__bf16)v.x; p.h[1] = (__bf16)v.y;
  p.h[2] = (__bf16)v.z; p.h[3] = (__bf16)v.w;
  *(uint2*)dst = p.u;                          // dst 8B aligned
}

// ---------------------------------------------------------------------------
// Kernel 1: router. One block per token group (b,n). 32x128 logits via bf16
// WMMA with a double-buffered, register-prefetched Wc K-chunk pipeline, then
// per-column argmax over the 32 tokens (tie toward larger token index,
// matching the reference's linspace tie-break).
// ---------------------------------------------------------------------------
__global__ __launch_bounds__(256) void router_kernel(
    const float* __restrict__ x, const float* __restrict__ Wc,
    int* __restrict__ tok) {
  __shared__ __align__(16) __bf16 Xl[G * DM];        // 64 KB [token][d]
  __shared__ __align__(16) __bf16 WcT[2][ESP * 32];  // 16 KB [es][k] x2 buffers
  __shared__ __align__(16) float  Lg[G * ESP];       // 16 KB logits [token][es]

  const int t  = threadIdx.x;
  const int gb = blockIdx.x;
  const int bb = gb / NGR;
  const int nn = gb % NGR;

  // Stage group tokens -> LDS bf16 (packed stores)
  const float4* xv = (const float4*)(x + (size_t)(bb * SS + nn * G) * DM);
#pragma unroll
  for (int i = 0; i < (G * DM / 4) / 256; ++i) {
    int lin = t + 256 * i;
    store_pk4(Xl + lin * 4, xv[lin]);
  }

  // Stage Wc chunk 0 transposed: WcT[es][k]  (Wc memory is [d][e*4+s]).
  // Each thread handles 8 (n, k-pair) cells -> packed b32 stores.
#pragma unroll
  for (int i = 0; i < 8; ++i) {
    int p = t + 256 * i;
    int n = p & 127;
    int k = (p >> 7) * 2;
    store_pk2(&WcT[0][n * 32 + k],
              Wc[(size_t)k * ESP + n], Wc[(size_t)(k + 1) * ESP + n]);
  }
  __syncthreads();

  const int wave = t >> 5;
  const int lane = t & 31;
  const int ln   = lane & 15;
  const int hi8  = (lane >= 16) ? 8 : 0;
  const int mt   = wave >> 2;        // M tile: 0..1 (32 tokens)
  const int nt0  = (wave & 3) * 2;   // N tiles: pairs over 128 es columns

  v8f acc0 = {}; v8f acc1 = {};
  float pa[8], pb[8];
  for (int kc = 0; kc < DM / 32; ++kc) {
    const int cur = kc & 1;
    // Prefetch next Wc chunk into registers (overlaps with WMMA below)
    if (kc < DM / 32 - 1) {
#pragma unroll
      for (int i = 0; i < 8; ++i) {
        int p = t + 256 * i;
        int n = p & 127;
        int k = (p >> 7) * 2;
        pa[i] = Wc[(size_t)((kc + 1) * 32 + k) * ESP + n];
        pb[i] = Wc[(size_t)((kc + 1) * 32 + k + 1) * ESP + n];
      }
    }
    v16bf a   = frag16(Xl + (mt * 16 + ln) * DM, kc * 32 + hi8);
    v16bf bf0 = frag16(&WcT[cur][(nt0 * 16 + ln) * 32], hi8);
    v16bf bf1 = frag16(&WcT[cur][((nt0 + 1) * 16 + ln) * 32], hi8);
    acc0 = wmma_bf16(a, bf0, acc0);
    acc1 = wmma_bf16(a, bf1, acc1);
    if (kc < DM / 32 - 1) {
#pragma unroll
      for (int i = 0; i < 8; ++i) {
        int p = t + 256 * i;
        int n = p & 127;
        int k = (p >> 7) * 2;
        store_pk2(&WcT[1 - cur][n * 32 + k], pa[i], pb[i]);
      }
    }
    __syncthreads();
  }

  // C layout: elem j -> row M = j (+8 for lanes>=16), col N = lane%16
#pragma unroll
  for (int j = 0; j < 8; ++j) {
    int row = mt * 16 + j + hi8;
    Lg[row * ESP + nt0 * 16 + ln]       = acc0[j];
    Lg[row * ESP + (nt0 + 1) * 16 + ln] = acc1[j];
  }
  __syncthreads();

  if (t < ESP) {
    // bc is constant per column -> doesn't change argmax. Tie term grows
    // with token index; ">=" scan makes the last max win, matching it.
    int best = 0;
    float bv = -3.4e38f;
    for (int r = 0; r < G; ++r) {
      float v = Lg[r * ESP + t] + (float)r * (1e-6f / 31.0f);
      if (v >= bv) { bv = v; best = r; }
    }
    tok[(size_t)gb * ESP + t] = best;
  }
}

// ---------------------------------------------------------------------------
// Kernel 2: per-(expert,set) MLP. Grid = 128 es * 4 row-blocks. Gathers 128
// selected token rows, runs relu(X*W1 + b1) * W2^T + b2 with bf16 WMMA
// (double-buffered, register-prefetched K pipeline for GEMM1), then
// atomically scatters the 128x1024 result back to the owning tokens.
// ---------------------------------------------------------------------------
__global__ __launch_bounds__(256) void expert_kernel(
    const float* __restrict__ x,
    const float* __restrict__ W1, const float* __restrict__ b1,
    const float* __restrict__ W2, const float* __restrict__ b2,
    const int* __restrict__ tok, float* __restrict__ out) {
  __shared__ __align__(16) __bf16 W2l[DM * 32];       // 64 KB [d][f]
  __shared__ __align__(16) __bf16 Ach[2][128 * 128];  // 64 KB gathered rows x2
  __shared__ __align__(16) __bf16 W1c[2][32 * 128];   // 16 KB [f][d_local] x2
  __shared__ __align__(16) __bf16 Inn[128 * 32];      //  8 KB hidden acts
  __shared__ __align__(16) float  b2l[DM];            //  4 KB
  __shared__ unsigned rowOff[128];                    // token-row offsets

  const int t  = threadIdx.x;
  const int es = blockIdx.x >> 2;   // 4 consecutive blocks share es -> L2 reuse
  const int mb = blockIdx.x & 3;

  const float* W1e = W1 + (size_t)es * DM * 32;
  const float* W2e = W2 + (size_t)es * DM * 32;

  // Stage W2 bf16 ([d][f] memory layout is already k=f-contiguous for GEMM2's
  // B fragment) and b2; compute gather==scatter row offsets.
  const float4* w2v = (const float4*)W2e;
#pragma unroll
  for (int i = 0; i < (DM * 32 / 4) / 256; ++i) {
    int lin = t + 256 * i;
    store_pk4(W2l + lin * 4, w2v[lin]);
  }
#pragma unroll
  for (int i = 0; i < DM / 256; ++i) {
    int lin = t + 256 * i;
    b2l[lin] = b2[(size_t)es * DM + lin];
  }
  if (t < 128) {
    int gb = mb * 128 + t;
    int bi = gb >> 8;      // / NGR
    int ni = gb & 255;
    int tk = tok[(size_t)gb * ESP + es];
    rowOff[t] = (unsigned)((bi * SS + ni * G + tk) * DM);
  }
  __syncthreads();   // rowOff visible before first gather

  const int wave = t >> 5;
  const int lane = t & 31;
  const int ln   = lane & 15;
  const int hi8  = (lane >= 16) ? 8 : 0;
  const int mt   = wave;            // 8 waves -> 8 M tiles of 16 rows
  const int row  = t >> 1;          // gather: 2 threads per row
  const int half = t & 1;

  // Stage chunk 0: gathered A rows + transposed W1 chunk
  {
    const float4* src = (const float4*)(x + (size_t)rowOff[row] + half * 64);
    __bf16* dst = Ach[0] + row * 128 + half * 64;
#pragma unroll
    for (int i = 0; i < 16; ++i) store_pk4(dst + i * 4, src[i]);
#pragma unroll
    for (int i = 0; i < 8; ++i) {
      int p = t + 256 * i;
      int f = p & 31;
      int dl = (p >> 5) * 2;
      store_pk2(&W1c[0][f * 128 + dl],
                W1e[(size_t)dl * 32 + f], W1e[(size_t)(dl + 1) * 32 + f]);
    }
  }
  __syncthreads();

  v8f acc0 = {}; v8f acc1 = {};
  float4 ga[16];
  float w1a[8], w1b[8];
  for (int kc = 0; kc < DM / 128; ++kc) {
    const int cur = kc & 1;
    // Prefetch next chunk (gather rows + W1 slab) into registers
    if (kc < DM / 128 - 1) {
      const float4* src =
          (const float4*)(x + (size_t)rowOff[row] + (kc + 1) * 128 + half * 64);
#pragma unroll
      for (int i = 0; i < 16; ++i) ga[i] = src[i];
#pragma unroll
      for (int i = 0; i < 8; ++i) {
        int p = t + 256 * i;
        int f = p & 31;
        int dg = (kc + 1) * 128 + (p >> 5) * 2;
        w1a[i] = W1e[(size_t)dg * 32 + f];
        w1b[i] = W1e[(size_t)(dg + 1) * 32 + f];
      }
    }
#pragma unroll
    for (int ks = 0; ks < 4; ++ks) {
      v16bf a  = frag16(Ach[cur] + (mt * 16 + ln) * 128, ks * 32 + hi8);
      v16bf bA = frag16(&W1c[cur][(0 * 16 + ln) * 128], ks * 32 + hi8);
      v16bf bB = frag16(&W1c[cur][(1 * 16 + ln) * 128], ks * 32 + hi8);
      acc0 = wmma_bf16(a, bA, acc0);
      acc1 = wmma_bf16(a, bB, acc1);
    }
    if (kc < DM / 128 - 1) {
      __bf16* dst = Ach[1 - cur] + row * 128 + half * 64;
#pragma unroll
      for (int i = 0; i < 16; ++i) store_pk4(dst + i * 4, ga[i]);
#pragma unroll
      for (int i = 0; i < 8; ++i) {
        int p = t + 256 * i;
        int f = p & 31;
        int dl = (p >> 5) * 2;
        store_pk2(&W1c[1 - cur][f * 128 + dl], w1a[i], w1b[i]);
      }
    }
    __syncthreads();
  }

  // bias + relu -> Inn (bf16)
#pragma unroll
  for (int j = 0; j < 8; ++j) {
    int rloc = mt * 16 + j + hi8;
    float v0 = acc0[j] + b1[(size_t)es * 32 + ln];
    float v1 = acc1[j] + b1[(size_t)es * 32 + 16 + ln];
    Inn[rloc * 32 + ln]      = (__bf16)(v0 > 0.f ? v0 : 0.f);
    Inn[rloc * 32 + 16 + ln] = (__bf16)(v1 > 0.f ? v1 : 0.f);
  }
  __syncthreads();

  // GEMM2: [128 x 32(f)] x [32(f) x 1024(d)], single K step per tile,
  // fused bias + atomic scatter back to token rows.
  v16bf a2 = frag16(Inn + (mt * 16 + ln) * 32, hi8);
  for (int nt = 0; nt < 64; ++nt) {
    v16bf bfrg = frag16(W2l + (nt * 16 + ln) * 32, hi8);
    v8f c = {};
    c = wmma_bf16(a2, bfrg, c);
    int col = nt * 16 + ln;
    float bias = b2l[col];
#pragma unroll
    for (int j = 0; j < 8; ++j) {
      int m = mt * 16 + j + hi8;
      atomicAdd(out + (size_t)rowOff[m] + col, c[j] + bias);
    }
  }
}

extern "C" void kernel_launch(void* const* d_in, const int* in_sizes, int n_in,
                              void* d_out, int out_size, void* d_ws,
                              size_t ws_size, hipStream_t stream) {
  (void)in_sizes; (void)n_in; (void)ws_size;
  const float* x  = (const float*)d_in[0];
  const float* Wc = (const float*)d_in[1];
  // d_in[2] = bc: constant per (e,s) column -> does not affect the argmax.
  const float* W1 = (const float*)d_in[3];
  const float* b1 = (const float*)d_in[4];
  const float* W2 = (const float*)d_in[5];
  const float* b2 = (const float*)d_in[6];
  float* out = (float*)d_out;
  int* tok = (int*)d_ws;  // NGROUPS * ESP ints = 256 KB

  hipMemsetAsync(out, 0, (size_t)out_size * sizeof(float), stream);
  router_kernel<<<NGROUPS, 256, 0, stream>>>(x, Wc, tok);
  expert_kernel<<<ESP * 4, 256, 0, stream>>>(x, W1, b1, W2, b2, tok, out);
}
